// minGRU_15668040696177
// MI455X (gfx1250) — matrix operand order repaired and checked by reading, not compiled, not run on hardware
//
#include <hip/hip_runtime.h>

// ---------------------------------------------------------------------------
// minGRU (Heinsen scan) for MI455X / gfx1250, wave32 + WMMA bf16.
//
//   K1 gemm_qg    : qg = x @ W^T, v_wmma_f32_16x16x32_bf16, 128x256x32 tiles,
//                   LDS double-buffered, fragments via ds_load_b128, staged
//                   operands converted to packed bf16 at load time,
//                   fused activation epilogue (max(q,EPS) / max(sigmoid,EPS))
//   K2 chunk_scan : per-chunk (256 steps) recurrence h=g*h+q, writes running
//                   gate product cumG in place over the Q buffer
//   K3 carry      : sequential combine of 16 chunk carries per channel
//   K4 fixup      : h = h_partial + carry[chunk] * cumG   (fully parallel)
// ---------------------------------------------------------------------------

typedef __attribute__((ext_vector_type(16))) __bf16 v16bf;
typedef __attribute__((ext_vector_type(8)))  float  v8f;

static constexpr int   Bb  = 4;
static constexpr int   Ss  = 4096;
static constexpr int   Dd  = 2048;          // model dim
static constexpr int   Mm  = Bb * Ss;       // 16384 rows
static constexpr int   Nn  = 2 * Dd;        // 4096 cols (query|gate)
static constexpr int   Kk  = Dd;            // 2048 reduction
static constexpr int   BM  = 128, BN = 256, BK = 32;
static constexpr int   LDT = 40;            // LDS row stride in halfs (80 B, 16B-aligned)
static constexpr int   NC  = 16;            // chunks over sequence
static constexpr int   CH  = Ss / NC;       // 256 steps per chunk
static constexpr float EPSV = 1e-20f;

__device__ __forceinline__ unsigned short f2bf(float f) {
  __bf16 h = (__bf16)f;                      // hardware f32->bf16 cvt (RNE)
  return __builtin_bit_cast(unsigned short, h);
}

__device__ __forceinline__ uint2 pack4(const float4& v) {
  uint2 r;
  r.x = f2bf(v.x) | ((unsigned)f2bf(v.y) << 16);
  r.y = f2bf(v.z) | ((unsigned)f2bf(v.w) << 16);
  return r;
}

// ---------------------------------------------------------------------------
// K1: qg = x @ W^T. Both operands are K-contiguous in memory.
// Per-wave 64x64 macro-tile: 16 WMMA per k-step vs 16 ds_load_b128.
// ---------------------------------------------------------------------------
__global__ __launch_bounds__(256) void gemm_qg(const float* __restrict__ X,
                                               const float* __restrict__ W,
                                               float* __restrict__ Q,
                                               float* __restrict__ G) {
  __shared__ unsigned short As[2][BM * LDT];   // bf16 bits, [m][k]
  __shared__ unsigned short Bs[2][BN * LDT];   // bf16 bits, [n][k]

  const int tid  = threadIdx.x;
  const int m0   = blockIdx.y * BM;
  const int n0   = blockIdx.x * BN;
  const int lane = tid & 31;
  const int wave = tid >> 5;
  const int wm   = wave & 1;                // 2 waves along M (64 rows each)
  const int wn   = wave >> 1;               // 4 waves along N (64 cols each)
  const int half = lane >> 4;               // lane group 0/1
  const int lr   = lane & 15;

  v8f acc[4][4] = {};

  const int lrow = tid >> 3;                // 0..31
  const int lcol = (tid & 7) * 4;           // 0,4,..,28

  // Staging: packed bf16 pairs, converted immediately after the global load
  // (24 VGPRs instead of 48 f32 -> keeps us inside the 256-VGPR window).
  uint2 ar[4], br[8];

  auto gload = [&](int k0) {
#pragma unroll
    for (int p = 0; p < 4; ++p)
      ar[p] = pack4(*reinterpret_cast<const float4*>(
          &X[(size_t)(m0 + lrow + 32 * p) * Kk + k0 + lcol]));
#pragma unroll
    for (int p = 0; p < 8; ++p)
      br[p] = pack4(*reinterpret_cast<const float4*>(
          &W[(size_t)(n0 + lrow + 32 * p) * Kk + k0 + lcol]));
  };
  auto lstore = [&](int buf) {
#pragma unroll
    for (int p = 0; p < 4; ++p)
      *reinterpret_cast<uint2*>(&As[buf][(lrow + 32 * p) * LDT + lcol]) = ar[p];
#pragma unroll
    for (int p = 0; p < 8; ++p)
      *reinterpret_cast<uint2*>(&Bs[buf][(lrow + 32 * p) * LDT + lcol]) = br[p];
  };

  union Frag { uint4 q[2]; v16bf v; };

  const int NT = Kk / BK;                   // 64 k-steps
  gload(0);
  lstore(0);
  __syncthreads();

#pragma unroll 1                            // keep live ranges < 256 VGPRs
  for (int kt = 0; kt < NT; ++kt) {
    const int cur = kt & 1;

    if (kt + 1 < NT) {
      gload((kt + 1) * BK);                 // issue next tile's HBM loads early
      if (kt + 2 < NT) {                    // CDNA5 global_prefetch_b8 at dist 2
        __builtin_prefetch(&X[(size_t)(m0 + lrow) * Kk + (kt + 2) * BK + lcol], 0, 1);
        __builtin_prefetch(&W[(size_t)(n0 + lrow) * Kk + (kt + 2) * BK + lcol], 0, 1);
      }
    }

    // A fragment (16x32 bf16): lane holds K in [8h,8h+8) and [16+8h,16+8h+8)
    Frag a[4];
#pragma unroll
    for (int i = 0; i < 4; ++i) {
      const unsigned short* pa =
          &As[cur][(wm * 64 + 16 * i + lr) * LDT + 8 * half];
      a[i].q[0] = *reinterpret_cast<const uint4*>(pa);
      a[i].q[1] = *reinterpret_cast<const uint4*>(pa + 16);
    }
    // B fragment (32x16 bf16): lane holds K in [8h,8h+8) and [16+8h,16+8h+8) at column lr
    Frag b[4];
#pragma unroll
    for (int j = 0; j < 4; ++j) {
      const unsigned short* pb =
          &Bs[cur][(wn * 64 + 16 * j + lr) * LDT + 8 * half];
      b[j].q[0] = *reinterpret_cast<const uint4*>(pb);
      b[j].q[1] = *reinterpret_cast<const uint4*>(pb + 16);
    }

#pragma unroll
    for (int i = 0; i < 4; ++i)
#pragma unroll
      for (int j = 0; j < 4; ++j)
        acc[i][j] = __builtin_amdgcn_wmma_f32_16x16x32_bf16(
            false, a[i].v, false, b[j].v, (short)0, acc[i][j], false, false);

    if (kt + 1 < NT) {
      lstore(1 - cur);                      // fill alternate buffer
      __syncthreads();                      // one barrier per k-step
    }
  }

  // Fused activation epilogue; N-tile is uniformly query or gate
  // (the 2048 boundary is a multiple of BN).
  const bool isQ = (n0 < Dd);
#pragma unroll
  for (int i = 0; i < 4; ++i) {
#pragma unroll
    for (int j = 0; j < 4; ++j) {
      const int n = n0 + wn * 64 + j * 16 + lr;
#pragma unroll
      for (int r = 0; r < 8; ++r) {
        const int m = m0 + wm * 64 + i * 16 + r + (half << 3);  // D: M=r(+8)
        const float v = acc[i][j][r];
        if (isQ) {
          Q[(size_t)m * Dd + n] = fmaxf(v, EPSV);
        } else {
          const float s = 1.0f / (1.0f + __expf(-v));
          G[(size_t)m * Dd + (n - Dd)] = fmaxf(s, EPSV);
        }
      }
    }
  }
}

// ---------------------------------------------------------------------------
// K2: per-chunk scan. One thread per (b, chunk, d); adjacent lanes take
// adjacent d -> coalesced. Writes partial h into H and the running gate
// product cumG in place over the Q buffer (strictly after reading q).
// ---------------------------------------------------------------------------
__global__ __launch_bounds__(256) void chunk_scan(float* __restrict__ QCG,
                                                  const float* __restrict__ G,
                                                  float* __restrict__ H) {
  const int idx = blockIdx.x * 256 + threadIdx.x;   // < B*D*NC = 131072
  const int d = idx & (Dd - 1);
  const int c = (idx >> 11) & (NC - 1);
  const int b = idx >> 15;
  size_t base = ((size_t)b * Ss + (size_t)c * CH) * Dd + d;

  float h = 0.0f, cg = 1.0f;
#pragma unroll 4
  for (int t = 0; t < CH; ++t) {
    const size_t i = base + (size_t)t * Dd;
    const float g = G[i];
    const float q = QCG[i];
    h  = fmaf(g, h, q);
    cg = cg * g;
    H[i]   = h;
    QCG[i] = cg;
  }
}

// ---------------------------------------------------------------------------
// K3: sequential carry combine across the 16 chunks per channel.
// carry[c] = final h entering chunk c.
// ---------------------------------------------------------------------------
__global__ __launch_bounds__(256) void carry_combine(const float* __restrict__ H,
                                                     const float* __restrict__ CG,
                                                     float* __restrict__ Carr) {
  const int idx = blockIdx.x * 256 + threadIdx.x;   // < B*D = 8192
  const int d = idx & (Dd - 1);
  const int b = idx >> 11;
  float carry = 0.0f;
#pragma unroll
  for (int c = 0; c < NC; ++c) {
    Carr[((size_t)b * NC + c) * Dd + d] = carry;
    const size_t ie = ((size_t)b * Ss + (size_t)c * CH + (CH - 1)) * Dd + d;
    carry = fmaf(CG[ie], carry, H[ie]);
  }
}

// ---------------------------------------------------------------------------
// K4: fully parallel fixup: h = h_partial + carry[chunk] * cumG.
// idx == ((b*S + t)*D + d) by construction.
// ---------------------------------------------------------------------------
__global__ __launch_bounds__(256) void fixup(float* __restrict__ H,
                                             const float* __restrict__ CG,
                                             const float* __restrict__ Carr) {
  const size_t idx = (size_t)blockIdx.x * 256 + threadIdx.x;  // < 33,554,432
  const int d = (int)(idx & (Dd - 1));
  const int t = (int)((idx >> 11) & (Ss - 1));
  const int b = (int)(idx >> 23);
  const int c = t >> 8;                                       // CH = 256
  const float carry = Carr[((size_t)b * NC + c) * Dd + d];
  H[idx] = fmaf(carry, CG[idx], H[idx]);
}

// ---------------------------------------------------------------------------
extern "C" void kernel_launch(void* const* d_in, const int* in_sizes, int n_in,
                              void* d_out, int out_size, void* d_ws, size_t ws_size,
                              hipStream_t stream) {
  const float* X = (const float*)d_in[0];   // [B,S,D]   f32
  const float* W = (const float*)d_in[1];   // [2D,D]    f32
  float* H = (float*)d_out;                 // [B,S,D]   f32

  // Workspace layout (floats):
  //   wsQ   [M*D]  : max(q,EPS); becomes cumG after K2 (in-place)
  //   wsG   [M*D]  : max(sigmoid(gate),EPS)
  //   wsCar [B*NC*D]
  float* wsQ   = (float*)d_ws;
  float* wsG   = wsQ + (size_t)Mm * Dd;
  float* wsCar = wsG + (size_t)Mm * Dd;

  dim3 g1(Nn / BN, Mm / BM);                           // 16 x 128 blocks
  gemm_qg<<<g1, 256, 0, stream>>>(X, W, wsQ, wsG);

  chunk_scan<<<(Bb * Dd * NC) / 256, 256, 0, stream>>>(wsQ, wsG, H);

  carry_combine<<<(Bb * Dd) / 256, 256, 0, stream>>>(H, wsQ, wsCar);

  fixup<<<(int)(((size_t)Mm * Dd) / 256), 256, 0, stream>>>(H, wsQ, wsCar);
}
